// SparseFinerAttention_65618510348704
// MI455X (gfx1250) — matrix operand order repaired and checked by reading, not compiled, not run on hardware
//
#include <hip/hip_runtime.h>

// ---------------------------------------------------------------------------
// SparseFinerAttention for MI455X (gfx1250, wave32, WMMA bf16 16x16x32)
//
// Pipeline (all bf16 matrix math, f32 accumulate, f32 softmax/topk math):
//  K0 cvt      : x, Wq, Wk, Wv, Wp  -> bf16 copies in workspace
//  K1 proj_qkv : Q = x@Wq^T, K = x@Wk^T  as [B,H,N,D] bf16
//                V = x@Wv^T stored TRANSPOSED as [B,H,D,N] bf16
//  K2 stats    : streaming QK^T (WMMA) + online softmax -> rowmax, rowsum.
//                K chunks staged cooperatively into LDS with
//                GLOBAL_LOAD_ASYNC_TO_LDS_B128 (double buffered, ASYNCcnt),
//                since all 8 waves of a block share the same (b,h) K slice.
//  K3 select+AV: per 16-row tile: (a) recompute logits, 1024-bin radix
//                histogram per row in LDS -> top-409 threshold bin + exact
//                remaining count; (b) recompute logits, write kept softmax
//                probs to LDS P[16][4096+pad] bf16; (c) out = P @ V via WMMA
//                from LDS, K-split over 8 waves, LDS f32 reduction ->
//                ctx [B,N,C] bf16
//  K4 out_proj : out = ctx@Wp^T + bp  (f32 output)
//
// Rationale: materializing attn (1.6GB f32) costs >=200us of HBM traffic at
// 23.3TB/s; recomputing QK^T on WMMA costs ~10us/pass and the 0.5MB K slice
// per (b,h) stays in the 192MB L2. Recompute wins by ~5x.
//
// Workspace requirement: ~69 MB.
// ---------------------------------------------------------------------------

#define DIM    768
#define NHEAD  12
#define HDIM   64
#define SEQ    4096
#define BATCH  2
#define ROWS   (BATCH*SEQ)      // 8192 token rows
#define BH     (BATCH*NHEAD)    // 24
#define KEEP   409              // int(4096*0.1)
#define ATTN_SCALE 0.125f       // 64^-0.5
#define NBINS  1024
#define PLD    (SEQ + 16)       // padded LDS row stride for P (bank-conflict)
#define KCH    64               // keys per async-staged chunk
#define KROW   72               // padded LDS row stride for K chunk (elems)
#define NCHUNK (SEQ / KCH)      // 64

typedef __attribute__((ext_vector_type(16))) __bf16 v16bf;
typedef __attribute__((ext_vector_type(8)))  float  v8f;

union Frag16 { unsigned int u32[8]; unsigned short u16[16]; v16bf v; };

__device__ __forceinline__ unsigned short f2bf(float f) {
  unsigned int u = __float_as_uint(f);
  u += 0x7fffu + ((u >> 16) & 1u);          // round-to-nearest-even
  return (unsigned short)(u >> 16);
}

// monotone key: larger float -> larger unsigned
__device__ __forceinline__ unsigned int fkey(float f) {
  unsigned int u = __float_as_uint(f);
  return (u & 0x80000000u) ? ~u : (u | 0x80000000u);
}

// A fragment 16x32 bf16: lane holds row M = lane%16.
// half=0 lanes: K 0..7 (v0..3), 16..23 (v4..7); half=1: K 8..15, 24..31.
// rowk points at &A[row][k0] (2-byte elems, k0 multiple of 32).
__device__ __forceinline__ v16bf load_a16(const unsigned short* rowk, int half) {
  const unsigned int* p = (const unsigned int*)rowk;
  Frag16 f;
#pragma unroll
  for (int j = 0; j < 4; ++j) f.u32[j] = p[4*half + j];
#pragma unroll
  for (int j = 0; j < 4; ++j) f.u32[4+j] = p[8 + 4*half + j];
  return f.v;
}

// B fragment 32x16 bf16: lane holds column N = lane%16; same K order as A:
// half=0: K 0..7, 16..23; half=1: K 8..15, 24..31. colk points at &B[col][k0].
__device__ __forceinline__ v16bf load_b16(const unsigned short* colk, int half) {
  const unsigned int* p = (const unsigned int*)colk;
  Frag16 f;
#pragma unroll
  for (int j = 0; j < 4; ++j) f.u32[j] = p[4*half + j];
#pragma unroll
  for (int j = 0; j < 4; ++j) f.u32[4+j] = p[8 + 4*half + j];
  return f.v;
}

__device__ __forceinline__ v8f wmma_bf16(v16bf a, v16bf b, v8f c) {
  return __builtin_amdgcn_wmma_f32_16x16x32_bf16(false, a, false, b,
                                                 (short)0, c, false, false);
}

// Async global->LDS copy, 16B per lane, tracked by ASYNCcnt (ISA 10.7/15.18).
__device__ __forceinline__ void async_ld_b128(unsigned int lds_off, const void* g) {
  asm volatile("global_load_async_to_lds_b128 %0, %1, off"
               :: "v"(lds_off), "v"((unsigned long long)(size_t)g)
               : "memory");
}

// ---------------------------------------------------------------- K0: convert
__global__ void cvt_bf16_kernel(const float* __restrict__ src,
                                unsigned short* __restrict__ dst, int n) {
  int i = blockIdx.x * blockDim.x + threadIdx.x;
  if (i < n) dst[i] = f2bf(src[i]);
}

// ---------------------------------------------------------------- K1: QKV proj
// grid (ROWS/16=512, 48/8=6, 3 projections), block 256 (8 waves).
// Each wave owns one 16x16 output tile; K-loop over 768 in steps of 32.
__global__ void __launch_bounds__(256)
proj_qkv_kernel(const unsigned short* __restrict__ xb,
                const unsigned short* __restrict__ Wqb,
                const unsigned short* __restrict__ Wkb,
                const unsigned short* __restrict__ Wvb,
                unsigned short* __restrict__ Q,
                unsigned short* __restrict__ Kd,
                unsigned short* __restrict__ Vt) {
  const int lane = threadIdx.x & 31, wave = threadIdx.x >> 5;
  const int r = lane & 15, half = lane >> 4;
  const int n0 = blockIdx.x * 16;
  const int ct = blockIdx.y * 8 + wave;
  const int c0 = ct * 16;
  const int mode = blockIdx.z;   // 0=Q 1=K 2=V
  const unsigned short* W = (mode == 0) ? Wqb : ((mode == 1) ? Wkb : Wvb);

  const unsigned short* arow = xb + (size_t)(n0 + r) * DIM;
  const unsigned short* bcol = W  + (size_t)(c0 + r) * DIM;   // y = x @ W^T
  v8f acc = {};
#pragma unroll 4
  for (int k0 = 0; k0 < DIM; k0 += 32) {
    v16bf a = load_a16(arow + k0, half);
    v16bf b = load_b16(bcol + k0, half);
    acc = wmma_bf16(a, b, acc);
  }

  const int h    = c0 / HDIM;        // r<16 and c0%64 in {0,16,32,48} -> same head
  const int dloc = (c0 & 63) + r;
#pragma unroll
  for (int j = 0; j < 8; ++j) {
    int n  = n0 + j + 8*half;        // D layout: VGPR j -> row j+8*half, col r
    int b_ = n >> 12, nin = n & 4095;
    size_t bh = (size_t)b_ * NHEAD + h;
    unsigned short val = f2bf(acc[j]);
    if (mode == 2) {
      Vt[(bh * HDIM + dloc) * SEQ + nin] = val;            // [B,H,D,N]
    } else {
      unsigned short* Out = (mode == 0) ? Q : Kd;
      Out[(bh * SEQ + nin) * HDIM + dloc] = val;           // [B,H,N,D]
    }
  }
}

// ---------------------------------------------------------------- K2: row stats
// 8 waves/block, one 16-query tile each, all sharing the same (b,h) slice.
// K streamed once per block through double-buffered async LDS chunks.
// Each chunk: 64 keys x 64 dims bf16 = 8KB, padded to 72-elem rows in LDS.
// Per wave per chunk: 2 async-load instructions -> ASYNCcnt discipline below.
__global__ void __launch_bounds__(256)
row_stats_kernel(const unsigned short* __restrict__ Q,
                 const unsigned short* __restrict__ Kd,
                 float* __restrict__ rowmax, float* __restrict__ rowsum) {
  __shared__ unsigned short kbuf[2][KCH * KROW];   // 2 x 9KB

  const int tid = threadIdx.x;
  const int lane = tid & 31, wave = tid >> 5;
  const int r = lane & 15, half = lane >> 4;
  const int tile = blockIdx.x * 8 + wave;        // 6144 tiles total
  const int bh = tile >> 8;                      // uniform across the block
  const int n0 = (tile & 255) * 16;

  const unsigned short* qrow = Q + ((size_t)bh * SEQ + n0 + r) * HDIM;
  v16bf a0 = load_a16(qrow, half);
  v16bf a1 = load_a16(qrow + 32, half);

  const unsigned short* ksrc = Kd + (size_t)bh * SEQ * HDIM;
  const int crow = tid >> 3, cseg = tid & 7;     // copy role: row 0..31, 16B seg

  // issue chunk c into kbuf[buf]: this thread copies rows crow and crow+32
  auto issue_chunk = [&](int c, int buf) {
    const unsigned short* g0 = ksrc + (size_t)c * KCH * HDIM
                                    + (size_t)crow * HDIM + cseg * 8;
    unsigned int l0 = (unsigned int)(size_t)&kbuf[buf][crow * KROW + cseg * 8];
    async_ld_b128(l0, g0);
    async_ld_b128(l0 + 32u * KROW * 2u, g0 + 32 * HDIM);
  };

  float m8[8], l8[8];
#pragma unroll
  for (int j = 0; j < 8; ++j) { m8[j] = -3.0e38f; l8[j] = 0.f; }

  issue_chunk(0, 0);
  issue_chunk(1, 1);
  for (int c = 0; c < NCHUNK; ++c) {
    // chunk c complete (async loads retire in order); chunk c+1 may be in flight
    if (c + 1 < NCHUNK) asm volatile("s_wait_asynccnt 0x2" ::: "memory");
    else                asm volatile("s_wait_asynccnt 0x0" ::: "memory");
    __syncthreads();                             // all waves' portions visible

    const unsigned short* kb = &kbuf[c & 1][0];
#pragma unroll
    for (int t = 0; t < KCH / 16; ++t) {
      const unsigned short* krow = kb + (size_t)(t * 16 + r) * KROW;
      v16bf b0 = load_b16(krow, half);
      v16bf b1 = load_b16(krow + 32, half);
      v8f d = {};
      d = wmma_bf16(a0, b0, d);
      d = wmma_bf16(a1, b1, d);
#pragma unroll
      for (int j = 0; j < 8; ++j) {
        float s = d[j] * ATTN_SCALE;
        float tmax = s;
#pragma unroll
        for (int off = 1; off < 16; off <<= 1)
          tmax = fmaxf(tmax, __shfl_xor(tmax, off, 16));
        float newm = fmaxf(m8[j], tmax);
        float tsum = __expf(s - newm);
#pragma unroll
        for (int off = 1; off < 16; off <<= 1)
          tsum += __shfl_xor(tsum, off, 16);
        l8[j] = l8[j] * __expf(m8[j] - newm) + tsum;
        m8[j] = newm;
      }
    }
    __syncthreads();                             // buffer free for reuse
    if (c + 2 < NCHUNK) issue_chunk(c + 2, c & 1);
  }

  if (r == 0) {                      // one lane per half writes its 8 rows
#pragma unroll
    for (int j = 0; j < 8; ++j) {
      size_t idx = (size_t)bh * SEQ + n0 + j + 8*half;
      rowmax[idx] = m8[j];
      rowsum[idx] = l8[j];
    }
  }
}

// ---------------------------------------------------------------- K3: select+AV
// One 256-thread block per 16-row tile (grid 6144). ~197KB dynamic LDS.
__global__ void __launch_bounds__(256)
select_av_kernel(const unsigned short* __restrict__ Q,
                 const unsigned short* __restrict__ Kd,
                 const unsigned short* __restrict__ Vt,
                 const float* __restrict__ rowmax,
                 const float* __restrict__ rowsum,
                 unsigned short* __restrict__ ctx) {
  extern __shared__ char smem[];
  unsigned short* P   = (unsigned short*)smem;                       // 16*PLD bf16
  unsigned int*  hist = (unsigned int*)(smem + 16*PLD*2);            // 16*1024 u32
  float*       outacc = (float*)(smem + 16*PLD*2 + 16*NBINS*4);      // 16*64 f32
  int*         thrbin = (int*)((char*)outacc + 16*HDIM*4);           // 16
  int*         remcnt = thrbin + 16;                                 // 16

  const int tid = threadIdx.x;
  const int lane = tid & 31, wave = tid >> 5;
  const int r = lane & 15, half = lane >> 4;
  const int bh = blockIdx.x >> 8;
  const int n0 = (blockIdx.x & 255) * 16;

  for (int i = tid; i < 16 * NBINS; i += 256) hist[i] = 0u;
  unsigned int* Pu = (unsigned int*)P;
  for (int i = tid; i < 16 * PLD / 2; i += 256) Pu[i] = 0u;
  for (int i = tid; i < 16 * HDIM; i += 256) outacc[i] = 0.f;
  __syncthreads();

  const unsigned short* qrow = Q + ((size_t)bh * SEQ + n0 + r) * HDIM;
  v16bf a0 = load_a16(qrow, half);
  v16bf a1 = load_a16(qrow + 32, half);
  const unsigned short* kbase = Kd + (size_t)bh * SEQ * HDIM;

  // ---- Phase A: radix histogram of logits (top 10 bits of order-key) ----
  for (int mt = wave; mt < SEQ / 16; mt += 8) {
    const unsigned short* krow = kbase + (size_t)(mt * 16 + r) * HDIM;
    __builtin_prefetch(krow + 128 * HDIM, 0, 1);         // global_prefetch_b8
    v16bf b0 = load_b16(krow, half);
    v16bf b1 = load_b16(krow + 32, half);
    v8f d = {};
    d = wmma_bf16(a0, b0, d);
    d = wmma_bf16(a1, b1, d);
#pragma unroll
    for (int j = 0; j < 8; ++j) {
      float s = d[j] * ATTN_SCALE;
      int row = j + 8*half;
      atomicAdd(&hist[row * NBINS + (fkey(s) >> 22)], 1u);
    }
  }
  __syncthreads();

  // ---- Phase B: threshold bin + exact count within bin ----
  if (tid < 16) {
    int row = tid, cum = 0, tb = 0, rem = KEEP;
    for (int bb = NBINS - 1; bb >= 0; --bb) {
      int c = (int)hist[row * NBINS + bb];
      if (cum + c >= KEEP) { tb = bb; rem = KEEP - cum; break; }
      cum += c;
    }
    thrbin[row] = tb;
    remcnt[row] = rem;
  }
  __syncthreads();

  float rm[8], ri[8];
#pragma unroll
  for (int j = 0; j < 8; ++j) {
    size_t idx = (size_t)bh * SEQ + n0 + j + 8*half;
    rm[j] = rowmax[idx];
    ri[j] = 1.0f / rowsum[idx];
  }

  // ---- Phase C: recompute logits, keep exactly KEEP probs per row ----
  for (int mt = wave; mt < SEQ / 16; mt += 8) {
    const unsigned short* krow = kbase + (size_t)(mt * 16 + r) * HDIM;
    v16bf b0 = load_b16(krow, half);
    v16bf b1 = load_b16(krow + 32, half);
    v8f d = {};
    d = wmma_bf16(a0, b0, d);
    d = wmma_bf16(a1, b1, d);
#pragma unroll
    for (int j = 0; j < 8; ++j) {
      float s = d[j] * ATTN_SCALE;
      int row = j + 8*half;
      int bin = (int)(fkey(s) >> 22);
      bool keep = bin > thrbin[row];
      if (bin == thrbin[row]) keep = atomicSub(&remcnt[row], 1) > 0;
      if (keep) {
        float p = __expf(s - rm[j]) * ri[j];
        P[row * PLD + mt * 16 + r] = f2bf(p);
      }
    }
  }
  __syncthreads();

  // ---- Phase D: out(16x64) = P(16x4096) @ V(4096x64), WMMA from LDS ----
  v8f acc[4] = {};
  for (int ks = wave; ks < SEQ / 32; ks += 8) {
    int k0 = ks * 32;
    v16bf a = load_a16(P + r * PLD + k0, half);
#pragma unroll
    for (int c = 0; c < 4; ++c) {
      const unsigned short* vcol = Vt + ((size_t)bh * HDIM + c*16 + r) * SEQ + k0;
      v16bf b = load_b16(vcol, half);
      acc[c] = wmma_bf16(a, b, acc[c]);
    }
  }
#pragma unroll
  for (int c = 0; c < 4; ++c)
#pragma unroll
    for (int j = 0; j < 8; ++j)
      atomicAdd(&outacc[(j + 8*half) * HDIM + c*16 + r], acc[c][j]);
  __syncthreads();

  const int b_ = bh / NHEAD, h = bh % NHEAD;
  for (int i = tid; i < 16 * HDIM; i += 256) {
    int row = i >> 6, dloc = i & 63;
    ctx[((size_t)b_ * SEQ + n0 + row) * DIM + h * HDIM + dloc] = f2bf(outacc[i]);
  }
}

// ---------------------------------------------------------------- K4: out proj
__global__ void __launch_bounds__(256)
out_proj_kernel(const unsigned short* __restrict__ ctx,
                const unsigned short* __restrict__ Wpb,
                const float* __restrict__ bp,
                float* __restrict__ out) {
  const int lane = threadIdx.x & 31, wave = threadIdx.x >> 5;
  const int r = lane & 15, half = lane >> 4;
  const int n0 = blockIdx.x * 16;
  const int c0 = (blockIdx.y * 8 + wave) * 16;

  const unsigned short* arow = ctx + (size_t)(n0 + r) * DIM;
  const unsigned short* bcol = Wpb + (size_t)(c0 + r) * DIM;
  v8f acc = {};
#pragma unroll 4
  for (int k0 = 0; k0 < DIM; k0 += 32) {
    v16bf a = load_a16(arow + k0, half);
    v16bf b = load_b16(bcol + k0, half);
    acc = wmma_bf16(a, b, acc);
  }
  float bias = bp[c0 + r];
#pragma unroll
  for (int j = 0; j < 8; ++j)
    out[(size_t)(n0 + j + 8*half) * DIM + c0 + r] = acc[j] + bias;
}

// ---------------------------------------------------------------- launch
extern "C" void kernel_launch(void* const* d_in, const int* in_sizes, int n_in,
                              void* d_out, int out_size, void* d_ws, size_t ws_size,
                              hipStream_t stream) {
  const float* x  = (const float*)d_in[0];
  const float* Wq = (const float*)d_in[1];
  const float* Wk = (const float*)d_in[2];
  const float* Wv = (const float*)d_in[3];
  const float* Wp = (const float*)d_in[4];
  const float* bp = (const float*)d_in[5];

  char* ws = (char*)d_ws;
  const size_t SZ_X  = (size_t)ROWS * DIM * 2;     // 12,582,912 B
  const size_t SZ_W  = (size_t)DIM * DIM * 2;      //  1,179,648 B
  const size_t SZ_P  = (size_t)BH * SEQ * HDIM * 2;// 12,582,912 B
  const size_t SZ_R  = (size_t)BH * SEQ * 4;       //    393,216 B

  size_t o = 0;
  unsigned short* xb  = (unsigned short*)(ws + o); o += SZ_X;
  unsigned short* Wqb = (unsigned short*)(ws + o); o += SZ_W;
  unsigned short* Wkb = (unsigned short*)(ws + o); o += SZ_W;
  unsigned short* Wvb = (unsigned short*)(ws + o); o += SZ_W;
  unsigned short* Wpb = (unsigned short*)(ws + o); o += SZ_W;
  unsigned short* Qp  = (unsigned short*)(ws + o); o += SZ_P;
  unsigned short* Kp  = (unsigned short*)(ws + o); o += SZ_P;
  unsigned short* Vtp = (unsigned short*)(ws + o); o += SZ_P;
  unsigned short* ctx = (unsigned short*)(ws + o); o += SZ_X;
  float* rmax = (float*)(ws + o); o += SZ_R;
  float* rsum = (float*)(ws + o); o += SZ_R;   // total ~68.4 MB required

  const int nx = ROWS * DIM, nw = DIM * DIM;
  cvt_bf16_kernel<<<(nx + 255) / 256, 256, 0, stream>>>(x,  xb,  nx);
  cvt_bf16_kernel<<<(nw + 255) / 256, 256, 0, stream>>>(Wq, Wqb, nw);
  cvt_bf16_kernel<<<(nw + 255) / 256, 256, 0, stream>>>(Wk, Wkb, nw);
  cvt_bf16_kernel<<<(nw + 255) / 256, 256, 0, stream>>>(Wv, Wvb, nw);
  cvt_bf16_kernel<<<(nw + 255) / 256, 256, 0, stream>>>(Wp, Wpb, nw);

  proj_qkv_kernel<<<dim3(ROWS / 16, 6, 3), 256, 0, stream>>>(
      xb, Wqb, Wkb, Wvb, Qp, Kp, Vtp);

  row_stats_kernel<<<dim3(BH * (SEQ / 16) / 8), 256, 0, stream>>>(
      Qp, Kp, rmax, rsum);

  const size_t smem = 16 * PLD * 2 + 16 * NBINS * 4 + 16 * HDIM * 4 + 32 * 4;
  select_av_kernel<<<dim3(BH * (SEQ / 16)), 256, smem, stream>>>(
      Qp, Kp, Vtp, rmax, rsum, ctx);

  out_proj_kernel<<<dim3(ROWS / 16, 6), 256, 0, stream>>>(
      ctx, Wpb, bp, (float*)d_out);
}
